// GCNLayer_60885456388981
// MI455X (gfx1250) — matrix operand order repaired and checked by reading, not compiled, not run on hardware
//
#include <hip/hip_runtime.h>

#define NN 16384   // graph nodes
#define IN_F 256
#define OUT_F 64

// ---- aggregation kernel tiling ----
#define ROWS   64                 // rows per block (4 waves x 16)
#define KC     128                // K-chunk: floats per row per TDM tile
#define PAD_DW 4                  // TDM LDS padding per row (dwords)
#define ROW_DW (KC + PAD_DW)      // 132 dwords
#define ROWB   (ROW_DW * 4)       // 528 bytes row pitch in LDS
#define BUFB   (ROWS * ROWB)      // 33792 bytes per ring slot
#define NBUF   4                  // ring depth (3 chunks in flight)
#define CHUNKS (NN / KC)          // 128

typedef __attribute__((ext_vector_type(16))) __bf16 v16bf;
typedef __attribute__((ext_vector_type(8)))  float  v8f;

#if __has_builtin(__builtin_amdgcn_tensor_load_to_lds)
#define HAS_TDM 1
typedef unsigned int u32x4 __attribute__((ext_vector_type(4)));
typedef int          i32x8 __attribute__((ext_vector_type(8)));
typedef int          i32x4 __attribute__((ext_vector_type(4)));

// Issue one TDM load: 2D tile [ROWS rows x KC floats] of adj -> LDS ring slot,
// with per-row LDS padding of PAD_DW dwords (bank-conflict mitigation).
// D# bit layout per CDNA5 ISA ch.8 (groups 0-3).
__device__ __forceinline__ void tdm_load_tile(const float* gptr, unsigned lds_off) {
    unsigned long long ga = (unsigned long long)(uintptr_t)gptr;
    u32x4 g0;
    g0[0] = 1u;                                          // count=1, user descriptor
    g0[1] = lds_off;                                     // lds_addr (bytes)
    g0[2] = (unsigned)(ga & 0xFFFFFFFFu);                // global_addr[31:0]
    g0[3] = (unsigned)((ga >> 32) & 0x01FFFFFFu)         // global_addr[56:32]
          | (2u << 30);                                  // type=2 ("image")
    i32x8 g1;
    g1[0] = (int)((2u << 16)                             // data_size = 4B
                | (1u << 20)                             // pad_enable
                | (6u << 22)                             // pad_interval: 2^6*8B = 128 dw
                | (3u << 25));                           // pad_amount: 4 dwords
    g1[1] = (int)(((unsigned)NN & 0xFFFFu) << 16);       // tensor_dim0[15:0]
    g1[2] = (int)((((unsigned)NN >> 16) & 0xFFFFu)       // tensor_dim0[31:16]
          | (((unsigned)NN & 0xFFFFu) << 16));           // tensor_dim1[15:0]
    g1[3] = (int)((((unsigned)NN >> 16) & 0xFFFFu)       // tensor_dim1[31:16]
          | ((unsigned)KC << 16));                       // tile_dim0 = KC
    g1[4] = (int)(unsigned)ROWS;                         // tile_dim1 = ROWS, tile_dim2=0
    g1[5] = (int)(unsigned)NN;                           // tensor_dim0_stride[31:0]
    g1[6] = 0;                                           // stride0[47:32]=0, stride1 lo=0
    g1[7] = 0;                                           // stride1 hi = 0
    i32x4 g2 = {0, 0, 0, 0};                             // no dims 2/3
    i32x4 g3 = {0, 0, 0, 0};
#if defined(__clang_major__) && (__clang_major__ >= 23)
    i32x8 g4 = {0, 0, 0, 0, 0, 0, 0, 0};                 // 6-arg toolchain variant
    __builtin_amdgcn_tensor_load_to_lds(g0, g1, g2, g3, g4, 0);
#else
    __builtin_amdgcn_tensor_load_to_lds(g0, g1, g2, g3, 0);
#endif
}
#else
#define HAS_TDM 0
#endif

// ---------------------------------------------------------------------------
// Kernel A: h = x @ W + b, stored TRANSPOSED as two bf16 planes (hi + lo
// residual) in workspace: hT_hi[col][k], hT_lo[col][k]. Tiny; VALU is fine.
// ---------------------------------------------------------------------------
__global__ __launch_bounds__(256) void gcn_linear_bf16split(
    const float* __restrict__ x, const float* __restrict__ W,
    const float* __restrict__ b,
    __bf16* __restrict__ hT_hi, __bf16* __restrict__ hT_lo)
{
    const int j = threadIdx.x & 63;
    const int i = blockIdx.x * 4 + (threadIdx.x >> 6);
    const float* __restrict__ xr = x + (size_t)i * IN_F;

    float acc = b[j];
#pragma unroll 8
    for (int k = 0; k < IN_F; ++k)
        acc = fmaf(xr[k], W[k * OUT_F + j], acc);

    __bf16 hi = (__bf16)acc;
    __bf16 lo = (__bf16)(acc - (float)hi);
    hT_hi[(size_t)j * NN + i] = hi;
    hT_lo[(size_t)j * NN + i] = lo;
}

// Prefetch all 8 B fragments (hi+lo planes, 4 column tiles) for K-step base KB
// into parity bank P of the ping-pong register file.
#define LOAD_B(P, KB) do {                                                    \
    const size_t kOff_ = (size_t)(KB) + (size_t)(lane_hi * 16);               \
    _Pragma("unroll")                                                         \
    for (int n_ = 0; n_ < 4; ++n_) {                                          \
        const size_t colOff_ = (size_t)(n_ * 16 + lane_lo) * NN + kOff_;      \
        bh[(P)][n_] = *(const v16bf*)(hT_hi + colOff_);                       \
        bl[(P)][n_] = *(const v16bf*)(hT_lo + colOff_);                       \
    }                                                                         \
} while (0)

// ---------------------------------------------------------------------------
// Kernel B: out = relu(adj @ h), split-precision bf16 WMMA.
//  - adj streamed HBM->LDS by the Tensor Data Mover, ring of NBUF slots,
//    issued NBUF-1 chunks ahead (256 blocks x 3 x 32KB ~ 24MB in flight to
//    cover 23.3 TB/s x HBM latency).
//  - B fragments (L2-resident transposed hT planes) and A LDS reads are
//    ping-pong double-buffered in registers one full K-step ahead: at
//    1 wave/SIMD structural occupancy, latency hiding must come from ILP.
//  - Per 32-K step: C += Ahi*Bhi + Ahi*Blo + Alo*Bhi (fp32-class accuracy),
//    12 v_wmma per K-step, 48 per chunk.
// ---------------------------------------------------------------------------
__global__ __launch_bounds__(128) void gcn_aggregate_wmma_tdm(
    const float*  __restrict__ adj,
    const __bf16* __restrict__ hT_hi,
    const __bf16* __restrict__ hT_lo,
    float*        __restrict__ out)
{
    extern __shared__ char smem[];   // NBUF * BUFB bytes

    const int lane    = threadIdx.x & 31;
    const int wave    = threadIdx.x >> 5;
    const int lane_lo = lane & 15;
    const int lane_hi = lane >> 4;
    const int rowBase = blockIdx.x * ROWS + wave * 16;

    const float* __restrict__ gBase = adj + (size_t)blockIdx.x * ROWS * NN;
    const unsigned smemOff = (unsigned)(uintptr_t)smem;  // LDS byte offset (addr[31:0])

    v8f c0 = {}, c1 = {}, c2 = {}, c3 = {};
    v16bf bh[2][4], bl[2][4];        // B ping-pong register file
    v8f a0[2], a1[2];                // A ping-pong (per-chunk)

#if HAS_TDM
    if (wave == 0) {
#pragma unroll
        for (int p = 0; p < NBUF - 1; ++p)
            tdm_load_tile(gBase + (size_t)p * KC, smemOff + (unsigned)(p * BUFB));
    }
#endif

    LOAD_B(0, 0);                    // B prologue for global K-step 0

    for (int c = 0; c < CHUNKS; ++c) {
        const int slot = c & (NBUF - 1);

#if HAS_TDM
        if (wave == 0) {
            const int nxt = c + (NBUF - 1);
            if (nxt < CHUNKS)
                tdm_load_tile(gBase + (size_t)nxt * KC,
                              smemOff + (unsigned)((nxt & (NBUF - 1)) * BUFB));
            const int rem = CHUNKS - 1 - c;   // newer chunks allowed outstanding
            if      (rem >= 3) __builtin_amdgcn_s_wait_tensorcnt(3);
            else if (rem == 2) __builtin_amdgcn_s_wait_tensorcnt(2);
            else if (rem == 1) __builtin_amdgcn_s_wait_tensorcnt(1);
            else               __builtin_amdgcn_s_wait_tensorcnt(0);
        }
#else
        {   // Fallback: cooperative global->LDS copy of chunk c.
            const int t = (int)threadIdx.x;       // 0..127
            const int r = t >> 1;                 // row 0..63
            const int hh = t & 1;                 // row half
            const float4* src = (const float4*)(gBase + (size_t)r * NN + (size_t)c * KC) + hh * 16;
            float4* dst = (float4*)(smem + slot * BUFB + r * ROWB) + hh * 16;
#pragma unroll
            for (int q = 0; q < 16; ++q) dst[q] = src[q];
        }
#endif
        __syncthreads();   // chunk c resident in LDS for all waves

        const float* __restrict__ aRowL =
            (const float*)(smem + slot * BUFB + (wave * 16 + lane_lo) * ROWB);

        // A prologue for this chunk (ks = 0)
        a0[0] = *(const v8f*)(aRowL + lane_hi * 16);
        a1[0] = *(const v8f*)(aRowL + 8 + lane_hi * 16);

#pragma unroll
        for (int ks = 0; ks < KC / 32; ++ks) {
            const int p = ks & 1;

            // Prefetch next K-step's A from LDS (stays within this chunk).
            if (ks < KC / 32 - 1) {
                a0[p ^ 1] = *(const v8f*)(aRowL + (ks + 1) * 32 + lane_hi * 16);
                a1[p ^ 1] = *(const v8f*)(aRowL + (ks + 1) * 32 + 8 + lane_hi * 16);
            }
            // Prefetch next global K-step's B (may cross the chunk barrier —
            // B is global; wrap keeps the dead last prefetch in-bounds).
            const int nkb = (c * KC + (ks + 1) * 32) & (NN - 1);
            LOAD_B(p ^ 1, nkb);

            // Split current A fragment into hi/lo bf16.
            v16bf ah, al;
#pragma unroll
            for (int e = 0; e < 8; ++e) {
                float f = a0[p][e];
                __bf16 h = (__bf16)f;
                ah[e] = h;
                al[e] = (__bf16)(f - (float)h);
            }
#pragma unroll
            for (int e = 0; e < 8; ++e) {
                float f = a1[p][e];
                __bf16 h = (__bf16)f;
                ah[8 + e] = h;
                al[8 + e] = (__bf16)(f - (float)h);
            }

#pragma unroll
            for (int n = 0; n < 4; ++n) {
                v8f* cc = (n == 0) ? &c0 : (n == 1) ? &c1 : (n == 2) ? &c2 : &c3;
                *cc = __builtin_amdgcn_wmma_f32_16x16x32_bf16(
                          false, ah, false, bh[p][n], (short)0, *cc, false, false);
                *cc = __builtin_amdgcn_wmma_f32_16x16x32_bf16(
                          false, ah, false, bl[p][n], (short)0, *cc, false, false);
                *cc = __builtin_amdgcn_wmma_f32_16x16x32_bf16(
                          false, al, false, bh[p][n], (short)0, *cc, false, false);
            }
        }
        __syncthreads();   // all waves done with slot before TDM reuses it
    }

    // Epilogue: ReLU + store. C/D layout: VGPR r, lanes 0-15 -> M=r,
    // lanes 16-31 -> M=r+8; column = lane_lo within the 16-col tile.
#pragma unroll
    for (int n = 0; n < 4; ++n) {
        const v8f* cc = (n == 0) ? &c0 : (n == 1) ? &c1 : (n == 2) ? &c2 : &c3;
#pragma unroll
        for (int r = 0; r < 8; ++r) {
            int row = rowBase + lane_hi * 8 + r;
            float v = (*cc)[r];
            out[(size_t)row * OUT_F + n * 16 + lane_lo] = v > 0.f ? v : 0.f;
        }
    }
}

// ---------------------------------------------------------------------------
extern "C" void kernel_launch(void* const* d_in, const int* in_sizes, int n_in,
                              void* d_out, int out_size, void* d_ws, size_t ws_size,
                              hipStream_t stream) {
    const float* x   = (const float*)d_in[0];   // [16384, 256]
    const float* adj = (const float*)d_in[1];   // [16384, 16384]
    const float* W   = (const float*)d_in[2];   // [256, 64]
    const float* b   = (const float*)d_in[3];   // [64]
    float* out = (float*)d_out;                 // [16384, 64]

    __bf16* hT_hi = (__bf16*)d_ws;
    __bf16* hT_lo = (__bf16*)((char*)d_ws + (size_t)OUT_F * NN * sizeof(__bf16));

    static const int smemBytes = NBUF * BUFB;   // 135168 B (< 320KB WGP LDS)
    (void)hipFuncSetAttribute((const void*)gcn_aggregate_wmma_tdm,
                              hipFuncAttributeMaxDynamicSharedMemorySize, smemBytes);

    gcn_linear_bf16split<<<NN / 4, 256, 0, stream>>>(x, W, b, hT_hi, hT_lo);
    gcn_aggregate_wmma_tdm<<<NN / ROWS, 128, smemBytes, stream>>>(adj, hT_hi, hT_lo, out);
}
